// TransformerBlock_11665131176261
// MI455X (gfx1250) — matrix operand order repaired and checked by reading, not compiled, not run on hardware
//
#include <hip/hip_runtime.h>

#define DEV __device__ __forceinline__

typedef __attribute__((ext_vector_type(16))) __bf16 v16bf;
typedef __attribute__((ext_vector_type(8)))  float  v8f;

// ---------- helpers ----------

DEV unsigned short f2bf(float f) {           // fp32 -> bf16, round-to-nearest-even
  union { float f; unsigned u; } x; x.f = f;
  unsigned r = x.u + 0x7FFFu + ((x.u >> 16) & 1u);
  return (unsigned short)(r >> 16);
}

// Assemble a 16-element bf16 fragment from two 16-byte chunks (LDS or global).
DEV v16bf frag2(const unsigned short* p0, const unsigned short* p1) {
  union { uint4 q[2]; v16bf v; } f;
  f.q[0] = *reinterpret_cast<const uint4*>(p0);
  f.q[1] = *reinterpret_cast<const uint4*>(p1);
  return f.v;
}

DEV v8f wmma_bf16(v16bf a, v16bf b, v8f c) {
  return __builtin_amdgcn_wmma_f32_16x16x32_bf16(
      false, a, false, b, (short)0, c, false, false);
}

// Async DMA: 16 bytes per lane, HBM -> LDS, tracked by ASYNCcnt.
// LDS dest address = low 32 bits of the generic pointer (aperture truncation).
DEV void async_b128(const unsigned short* gsrc, unsigned short* ldst) {
  unsigned l = (unsigned)(size_t)ldst;
  unsigned long long g = (unsigned long long)(size_t)gsrc;
  asm volatile("global_load_async_to_lds_b128 %0, %1, off"
               :: "v"(l), "v"(g) : "memory");
}

DEV void wait_async0() {
  asm volatile("s_wait_asynccnt 0x0" ::: "memory");
}

// ---------- fp32 -> bf16 bulk convert (weights) ----------

__global__ void f32_to_bf16_kernel(const float* __restrict__ in,
                                   unsigned short* __restrict__ out, int n) {
  int i = blockIdx.x * blockDim.x + threadIdx.x;
  int stride = gridDim.x * blockDim.x;
  for (; i < n; i += stride) out[i] = f2bf(in[i]);
}

// ---------- LayerNorm (D=1024, 256 threads, one row per block) ----------

__global__ __launch_bounds__(256)
void layernorm_bf16_kernel(const float* __restrict__ x,
                           const float* __restrict__ gamma,
                           unsigned short* __restrict__ out) {
  const int D = 1024;
  const int row = blockIdx.x;
  const float4 v = reinterpret_cast<const float4*>(x + (size_t)row * D)[threadIdx.x];

  float s = v.x + v.y + v.z + v.w;
  s += __shfl_xor(s, 1);  s += __shfl_xor(s, 2);
  s += __shfl_xor(s, 4);  s += __shfl_xor(s, 8);  s += __shfl_xor(s, 16);

  __shared__ float red[8];
  const int w = threadIdx.x >> 5, lane = threadIdx.x & 31;
  if (lane == 0) red[w] = s;
  __syncthreads();
  float tot = 0.f;
#pragma unroll
  for (int i = 0; i < 8; ++i) tot += red[i];
  const float mu = tot * (1.f / D);

  float dx = v.x - mu, dy = v.y - mu, dz = v.z - mu, dw = v.w - mu;
  float s2 = dx * dx + dy * dy + dz * dz + dw * dw;
  s2 += __shfl_xor(s2, 1);  s2 += __shfl_xor(s2, 2);
  s2 += __shfl_xor(s2, 4);  s2 += __shfl_xor(s2, 8);  s2 += __shfl_xor(s2, 16);
  __syncthreads();
  if (lane == 0) red[w] = s2;
  __syncthreads();
  float tot2 = 0.f;
#pragma unroll
  for (int i = 0; i < 8; ++i) tot2 += red[i];
  const float rstd = rsqrtf(tot2 * (1.f / D) + 1e-5f);

  const float4 g = reinterpret_cast<const float4*>(gamma)[threadIdx.x];
  unsigned short* o = out + (size_t)row * D + threadIdx.x * 4;
  o[0] = f2bf(dx * rstd * g.x);
  o[1] = f2bf(dy * rstd * g.y);
  o[2] = f2bf(dz * rstd * g.z);
  o[3] = f2bf(dw * rstd * g.w);
}

// ---------- generic bf16 WMMA GEMM:  C[M,N] = A[M,K] * B[N,K]^T (+epilogue) ----------
// 256 threads = 8 waves, 128x128 tile, BK=32, double-buffered async HBM->LDS DMA.
// Wave tile 32x64 (2x4 v8f accumulators).
// EPI: 0 = store f32, 1 = +resid, 2 = +bias[n]+resid

template <int EPI>
__global__ __launch_bounds__(256)
void gemm_bf16_kernel(const unsigned short* __restrict__ A,
                      const unsigned short* __restrict__ B,
                      float* __restrict__ C,
                      const float* __restrict__ resid,
                      const float* __restrict__ bias,
                      int K, int lda, int ldb, int ldc) {
  __shared__ unsigned short As[2][128 * 40];
  __shared__ unsigned short Bs[2][128 * 40];

  const int tid  = threadIdx.x;
  const int lane = tid & 31;
  const int w    = tid >> 5;
  const int l16  = lane & 15;
  const int half = lane >> 4;
  const int wm   = (w >> 1) * 32;
  const int wn   = (w & 1) * 64;
  const int m0   = blockIdx.y * 128;
  const int n0   = blockIdx.x * 128;

  const int trow = tid >> 1;
  const int tcol = (tid & 1) * 16;
  const unsigned short* gA = A + (size_t)(m0 + trow) * lda + tcol;
  const unsigned short* gB = B + (size_t)(n0 + trow) * ldb + tcol;
  const int lofs = trow * 40 + tcol;

  v8f acc[2][4] = {};

  // prologue: stage K-tile 0 into buffer 0
#pragma unroll
  for (int e = 0; e < 2; ++e) {
    async_b128(gA + e * 8, &As[0][lofs + e * 8]);
    async_b128(gB + e * 8, &Bs[0][lofs + e * 8]);
  }

  int buf = 0;
  for (int ko = 0; ko < K; ko += 32) {
    wait_async0();                 // this wave's DMA into buf complete
    __syncthreads();               // everyone's DMA complete / prev reads done
    if (ko + 32 < K) {             // kick off next tile into the other buffer
#pragma unroll
      for (int e = 0; e < 2; ++e) {
        async_b128(gA + ko + 32 + e * 8, &As[buf ^ 1][lofs + e * 8]);
        async_b128(gB + ko + 32 + e * 8, &Bs[buf ^ 1][lofs + e * 8]);
      }
    }

    v16bf af[2], bf[4];
#pragma unroll
    for (int i = 0; i < 2; ++i) {
      const unsigned short* p = &As[buf][(wm + i * 16 + l16) * 40 + half * 8];
      af[i] = frag2(p, p + 16);    // A: K-chunks {0..7,16..23}/{8..15,24..31}
    }
#pragma unroll
    for (int j = 0; j < 4; ++j) {
      const unsigned short* p = &Bs[buf][(wn + j * 16 + l16) * 40 + half * 8];
      bf[j] = frag2(p, p + 16);    // B: K-chunks {0..7,16..23}/{8..15,24..31}
    }
#pragma unroll
    for (int i = 0; i < 2; ++i)
#pragma unroll
      for (int j = 0; j < 4; ++j)
        acc[i][j] = wmma_bf16(af[i], bf[j], acc[i][j]);

    buf ^= 1;
  }

#pragma unroll
  for (int i = 0; i < 2; ++i) {
#pragma unroll
    for (int j = 0; j < 4; ++j) {
#pragma unroll
      for (int r = 0; r < 8; ++r) {
        const int gm = m0 + wm + i * 16 + half * 8 + r;
        const int gn = n0 + wn + j * 16 + l16;
        const size_t idx = (size_t)gm * ldc + gn;
        float v = acc[i][j][r];
        if (EPI == 1) v += resid[idx];
        if (EPI == 2) v += bias[gn] + resid[idx];
        C[idx] = v;
      }
    }
  }
}

// ---------- RoPE + split into head-major bf16 Q/K/V ----------
// qkv fp32 [B,S,3*D]; out [B,H,S,64] bf16

__global__ void rope_split_kernel(const float* __restrict__ qkv,
                                  const float* __restrict__ freqs,
                                  unsigned short* __restrict__ Qh,
                                  unsigned short* __restrict__ Kh,
                                  unsigned short* __restrict__ Vh) {
  const int S = 2048, H = 16;
  int i = blockIdx.x * blockDim.x + threadIdx.x;   // B*S*H*64 threads
  const int d = i & 63;
  int t = i >> 6;
  const int h = t & 15;  t >>= 4;
  const int s = t & 2047;
  const int b = t >> 11;

  const size_t src = ((size_t)(b * S + s)) * 3072 + h * 64 + d;
  float q = qkv[src], k = qkv[src + 1024], v = qkv[src + 2048];
  if (d < 32) {
    const float f = freqs[s * 32 + d];
    const float c = __cosf(f), sn = __sinf(f);
    const int po   = (d < 16) ? 16 : -16;
    const float sg = (d < 16) ? -1.f : 1.f;
    q = q * c + sg * qkv[src + po] * sn;
    k = k * c + sg * qkv[src + 1024 + po] * sn;
  }
  const size_t dst = ((size_t)((b * H + h) * S + s)) * 64 + d;
  Qh[dst] = f2bf(q);  Kh[dst] = f2bf(k);  Vh[dst] = f2bf(v);
}

// ---------- causal flash attention (bf16 WMMA, fp32 online softmax) ----------
// 128 threads = 4 waves; block handles (b*H+h, 64 query rows); wave owns 16 rows.

__global__ __launch_bounds__(128)
void attn_kernel(const unsigned short* __restrict__ Q,
                 const unsigned short* __restrict__ K,
                 const unsigned short* __restrict__ V,
                 unsigned short* __restrict__ ctx) {
  const int S = 2048, H = 16, DH = 64;
  __shared__ unsigned short Ks[32 * 72];      // [j][dh]
  __shared__ unsigned short Vs[64 * 40];      // transposed: [dh][j]
  __shared__ unsigned short Ps[4][16 * 40];   // per-wave P tile (C->A relayout)

  const int tid = threadIdx.x;
  const int lane = tid & 31, w = tid >> 5;
  const int l16 = lane & 15, half = lane >> 4;
  const int bh = blockIdx.y;
  const int qBase = blockIdx.x * 64;
  const int qRow0 = qBase + w * 16;

  const size_t hb = (size_t)bh * S * DH;
  const unsigned short* Qp = Q + hb;
  const unsigned short* Kp = K + hb;
  const unsigned short* Vp = V + hb;

  v16bf qa[2];
  {
    const unsigned short* qr = Qp + (size_t)(qRow0 + l16) * DH;
#pragma unroll
    for (int c = 0; c < 2; ++c) {
      const unsigned short* p = qr + c * 32 + half * 8;
      qa[c] = frag2(p, p + 16);
    }
  }

  float mstat[8], lstat[8];
#pragma unroll
  for (int r = 0; r < 8; ++r) { mstat[r] = -1e30f; lstat[r] = 0.f; }
  v8f acc[4] = {};
  const float scale = 0.125f;                 // 1/sqrt(64)

  const int jMax = qBase + 64;
  const int srow = tid >> 2, sseg = (tid & 3) * 16;

  for (int j0 = 0; j0 < jMax; j0 += 32) {
    __syncthreads();
    {   // K tile via async DMA (row-major)
      const unsigned short* gk = Kp + (size_t)(j0 + srow) * DH + sseg;
      async_b128(gk,     &Ks[srow * 72 + sseg]);
      async_b128(gk + 8, &Ks[srow * 72 + sseg + 8]);
      // V tile transposed manually
      union { uint4 q[2]; unsigned short us[16]; } vv;
      const uint4* gv = reinterpret_cast<const uint4*>(Vp + (size_t)(j0 + srow) * DH + sseg);
      vv.q[0] = gv[0]; vv.q[1] = gv[1];
#pragma unroll
      for (int e = 0; e < 16; ++e) Vs[(sseg + e) * 40 + srow] = vv.us[e];
    }
    wait_async0();
    __syncthreads();

    if (j0 < qRow0 + 16) {
      v8f sc[2];
#pragma unroll
      for (int js = 0; js < 2; ++js) {
        v8f s = {};
#pragma unroll
        for (int c = 0; c < 2; ++c) {
          const unsigned short* p = &Ks[(js * 16 + l16) * 72 + c * 32 + half * 8];
          s = wmma_bf16(qa[c], frag2(p, p + 16), s);
        }
#pragma unroll
        for (int r = 0; r < 8; ++r) {
          const int row = qRow0 + half * 8 + r;
          const int col = j0 + js * 16 + l16;
          const float v = s[r] * scale;
          s[r] = (col <= row) ? v : -1e30f;
        }
        sc[js] = s;
      }

      float mnew[8], alpha[8], rsum[8];
#pragma unroll
      for (int r = 0; r < 8; ++r) {
        float mx = fmaxf(sc[0][r], sc[1][r]);
        mx = fmaxf(mx, __shfl_xor(mx, 1));
        mx = fmaxf(mx, __shfl_xor(mx, 2));
        mx = fmaxf(mx, __shfl_xor(mx, 4));
        mx = fmaxf(mx, __shfl_xor(mx, 8));     // half-wave row reduction
        mnew[r]  = fmaxf(mstat[r], mx);
        alpha[r] = __expf(mstat[r] - mnew[r]);
        mstat[r] = mnew[r];
        rsum[r]  = 0.f;
      }
#pragma unroll
      for (int js = 0; js < 2; ++js)
#pragma unroll
        for (int r = 0; r < 8; ++r) {
          const float p = __expf(sc[js][r] - mnew[r]);
          rsum[r] += p;
          Ps[w][(half * 8 + r) * 40 + js * 16 + l16] = f2bf(p);
        }
#pragma unroll
      for (int r = 0; r < 8; ++r) {
        float s = rsum[r];
        s += __shfl_xor(s, 1);  s += __shfl_xor(s, 2);
        s += __shfl_xor(s, 4);  s += __shfl_xor(s, 8);
        lstat[r] = lstat[r] * alpha[r] + s;
      }
#pragma unroll
      for (int n = 0; n < 4; ++n)
#pragma unroll
        for (int r = 0; r < 8; ++r) acc[n][r] *= alpha[r];

      const unsigned short* pp = &Ps[w][l16 * 40 + half * 8];
      const v16bf pa = frag2(pp, pp + 16);          // A-layout P fragment
#pragma unroll
      for (int n = 0; n < 4; ++n) {
        const unsigned short* vp2 = &Vs[(n * 16 + l16) * 40 + half * 8];
        acc[n] = wmma_bf16(pa, frag2(vp2, vp2 + 16), acc[n]);
      }
    }
  }

  const int b = bh / H, h = bh % H;
#pragma unroll
  for (int n = 0; n < 4; ++n)
#pragma unroll
    for (int r = 0; r < 8; ++r) {
      const int row = qRow0 + half * 8 + r;
      const int col = h * 64 + n * 16 + l16;
      const float v = acc[n][r] / lstat[r];
      ctx[((size_t)(b * S + row)) * 1024 + col] = f2bf(v);
    }
}

// ---------- FF1 dual-GEMM + fused SwiGLU -> bf16 ----------
// hid = xn2*W1[n,:], gate = xn2*W1[4096+n,:], ff = (hid+b)*silu(gate+b)
// Double-buffered async staging of A and both weight streams.

__global__ __launch_bounds__(256)
void ff1_swiglu_kernel(const unsigned short* __restrict__ A,   // [4096,1024] bf16
                       const unsigned short* __restrict__ W,   // [8192,1024] bf16
                       const float* __restrict__ bias,         // [8192]
                       unsigned short* __restrict__ out) {     // [4096,4096] bf16
  const int K = 1024, lda = 1024, ldb = 1024, INNER = 4096;
  __shared__ unsigned short As[2][128 * 40];
  __shared__ unsigned short Bh[2][128 * 40];
  __shared__ unsigned short Bg[2][128 * 40];

  const int tid = threadIdx.x;
  const int lane = tid & 31, w = tid >> 5;
  const int l16 = lane & 15, half = lane >> 4;
  const int wm = (w >> 1) * 32, wn = (w & 1) * 64;
  const int m0 = blockIdx.y * 128, n0 = blockIdx.x * 128;
  const int trow = tid >> 1, tcol = (tid & 1) * 16;

  const unsigned short* gA = A + (size_t)(m0 + trow) * lda + tcol;
  const unsigned short* gH = W + (size_t)(n0 + trow) * ldb + tcol;
  const unsigned short* gG = W + (size_t)(INNER + n0 + trow) * ldb + tcol;
  const int lofs = trow * 40 + tcol;

  v8f ah[2][4] = {}, ag[2][4] = {};

#pragma unroll
  for (int e = 0; e < 2; ++e) {
    async_b128(gA + e * 8, &As[0][lofs + e * 8]);
    async_b128(gH + e * 8, &Bh[0][lofs + e * 8]);
    async_b128(gG + e * 8, &Bg[0][lofs + e * 8]);
  }

  int buf = 0;
  for (int ko = 0; ko < K; ko += 32) {
    wait_async0();
    __syncthreads();
    if (ko + 32 < K) {
#pragma unroll
      for (int e = 0; e < 2; ++e) {
        async_b128(gA + ko + 32 + e * 8, &As[buf ^ 1][lofs + e * 8]);
        async_b128(gH + ko + 32 + e * 8, &Bh[buf ^ 1][lofs + e * 8]);
        async_b128(gG + ko + 32 + e * 8, &Bg[buf ^ 1][lofs + e * 8]);
      }
    }

    v16bf af[2], bfh[4], bfg[4];
#pragma unroll
    for (int i = 0; i < 2; ++i) {
      const unsigned short* p = &As[buf][(wm + i * 16 + l16) * 40 + half * 8];
      af[i] = frag2(p, p + 16);
    }
#pragma unroll
    for (int j = 0; j < 4; ++j) {
      const unsigned short* ph = &Bh[buf][(wn + j * 16 + l16) * 40 + half * 8];
      const unsigned short* pg = &Bg[buf][(wn + j * 16 + l16) * 40 + half * 8];
      bfh[j] = frag2(ph, ph + 16);
      bfg[j] = frag2(pg, pg + 16);
    }
#pragma unroll
    for (int i = 0; i < 2; ++i)
#pragma unroll
      for (int j = 0; j < 4; ++j) {
        ah[i][j] = wmma_bf16(af[i], bfh[j], ah[i][j]);
        ag[i][j] = wmma_bf16(af[i], bfg[j], ag[i][j]);
      }
    buf ^= 1;
  }

#pragma unroll
  for (int i = 0; i < 2; ++i)
#pragma unroll
    for (int j = 0; j < 4; ++j)
#pragma unroll
      for (int r = 0; r < 8; ++r) {
        const int gm = m0 + wm + i * 16 + half * 8 + r;
        const int gn = n0 + wn + j * 16 + l16;
        const float hv = ah[i][j][r] + bias[gn];
        const float gv = ag[i][j][r] + bias[INNER + gn];
        const float ff = hv * (gv / (1.f + __expf(-gv)));
        out[(size_t)gm * INNER + gn] = f2bf(ff);
      }
}

// ---------- host launch ----------

extern "C" void kernel_launch(void* const* d_in, const int* in_sizes, int n_in,
                              void* d_out, int out_size, void* d_ws, size_t ws_size,
                              hipStream_t stream) {
  (void)in_sizes; (void)n_in; (void)out_size; (void)ws_size;
  const int B = 2, S = 2048, D = 1024, H = 16, DH = 64;
  const int M = B * S, INNER = 4 * D;

  const float* x      = (const float*)d_in[0];
  const float* freqs  = (const float*)d_in[1];
  const float* W_qkv  = (const float*)d_in[2];
  const float* W_out  = (const float*)d_in[3];
  const float* gamma1 = (const float*)d_in[4];
  const float* gamma2 = (const float*)d_in[5];
  const float* W_ff1  = (const float*)d_in[6];
  const float* b_ff1  = (const float*)d_in[7];
  const float* W_ff2  = (const float*)d_in[8];
  const float* b_ff2  = (const float*)d_in[9];
  float* out = (float*)d_out;

  char* ws = (char*)d_ws;
  size_t off = 0;
  auto alloc = [&](size_t bytes) -> void* {
    void* p = ws + off;
    off = (off + bytes + 255) & ~(size_t)255;
    return p;
  };
  unsigned short* wqkv_bf = (unsigned short*)alloc((size_t)3 * D * D * 2);
  unsigned short* wout_bf = (unsigned short*)alloc((size_t)D * D * 2);
  unsigned short* wff1_bf = (unsigned short*)alloc((size_t)2 * INNER * D * 2);
  unsigned short* wff2_bf = (unsigned short*)alloc((size_t)D * INNER * 2);
  unsigned short* xn_bf   = (unsigned short*)alloc((size_t)M * D * 2);
  float*          qkv_f32 = (float*)alloc((size_t)M * 3 * D * 4);
  unsigned short* q_bf    = (unsigned short*)alloc((size_t)M * D * 2);
  unsigned short* k_bf    = (unsigned short*)alloc((size_t)M * D * 2);
  unsigned short* v_bf    = (unsigned short*)alloc((size_t)M * D * 2);
  unsigned short* ctx_bf  = (unsigned short*)alloc((size_t)M * D * 2);
  float*          x1_f32  = (float*)alloc((size_t)M * D * 4);
  unsigned short* xn2_bf  = (unsigned short*)alloc((size_t)M * D * 2);
  unsigned short* ff_bf   = (unsigned short*)alloc((size_t)M * INNER * 2);

  // 1) weights -> bf16
  f32_to_bf16_kernel<<<2048, 256, 0, stream>>>(W_qkv, wqkv_bf, 3 * D * D);
  f32_to_bf16_kernel<<<2048, 256, 0, stream>>>(W_out, wout_bf, D * D);
  f32_to_bf16_kernel<<<2048, 256, 0, stream>>>(W_ff1, wff1_bf, 2 * INNER * D);
  f32_to_bf16_kernel<<<2048, 256, 0, stream>>>(W_ff2, wff2_bf, D * INNER);

  // 2) LN1 -> bf16
  layernorm_bf16_kernel<<<M, 256, 0, stream>>>(x, gamma1, xn_bf);

  // 3) QKV GEMM (fp32 out for RoPE precision)
  gemm_bf16_kernel<0><<<dim3(3 * D / 128, M / 128), 256, 0, stream>>>(
      xn_bf, wqkv_bf, qkv_f32, nullptr, nullptr, D, D, D, 3 * D);

  // 4) RoPE + split into head-major bf16
  rope_split_kernel<<<(B * S * H * DH) / 256, 256, 0, stream>>>(
      qkv_f32, freqs, q_bf, k_bf, v_bf);

  // 5) causal flash attention -> ctx bf16 [B,S,D]
  attn_kernel<<<dim3(S / 64, B * H), 128, 0, stream>>>(q_bf, k_bf, v_bf, ctx_bf);

  // 6) output projection + residual -> x1 fp32
  gemm_bf16_kernel<1><<<dim3(D / 128, M / 128), 256, 0, stream>>>(
      ctx_bf, wout_bf, x1_f32, x, nullptr, D, D, D, D);

  // 7) LN2 -> bf16
  layernorm_bf16_kernel<<<M, 256, 0, stream>>>(x1_f32, gamma2, xn2_bf);

  // 8) FF1 dual GEMM + SwiGLU -> ff bf16
  ff1_swiglu_kernel<<<dim3(INNER / 128, M / 128), 256, 0, stream>>>(
      xn2_bf, wff1_bf, b_ff1, ff_bf);

  // 9) FF2 GEMM + bias + residual -> out
  gemm_bf16_kernel<2><<<dim3(D / 128, M / 128), 256, 0, stream>>>(
      ff_bf, wff2_bf, out, x1_f32, b_ff2, INNER, INNER, INNER, D);
}
